// SelfAttention_59700045414961
// MI455X (gfx1250) — matrix operand order repaired and checked by reading, not compiled, not run on hardware
//
#include <hip/hip_runtime.h>

typedef __attribute__((ext_vector_type(16))) _Float16 v16h;
typedef __attribute__((ext_vector_type(8)))  _Float16 v8h;
typedef __attribute__((ext_vector_type(8)))  float    v8f;

#define B_   16
#define C_   256
#define W_   64
#define N_   4096
#define N4_  1024
#define CQ_  32
#define CV_  128

union AV { v16h v; v8h h[2]; };

static __device__ __forceinline__ v8f wmma_f16(const AV& a, const AV& b, v8f c) {
  return __builtin_amdgcn_wmma_f32_16x16x32_f16(false, a.v, false, b.v,
                                                (short)0, c, false, false);
}

// Max-reduce across each 16-lane row group using DPP16 row ops (VALU only,
// no LDS traffic; co-executes with the XDL/WMMA pipe).
static __device__ __forceinline__ float wave16_max(float x) {
  int xi, yi;
  xi = __builtin_bit_cast(int, x);
  yi = __builtin_amdgcn_update_dpp(0, xi, 0xB1, 0xF, 0xF, true); // quad xor1
  x  = fmaxf(x, __builtin_bit_cast(float, yi));
  xi = __builtin_bit_cast(int, x);
  yi = __builtin_amdgcn_update_dpp(0, xi, 0x4E, 0xF, 0xF, true); // quad xor2
  x  = fmaxf(x, __builtin_bit_cast(float, yi));
  xi = __builtin_bit_cast(int, x);
  yi = __builtin_amdgcn_update_dpp(0, xi, 0x141, 0xF, 0xF, true); // half mirror
  x  = fmaxf(x, __builtin_bit_cast(float, yi));
  xi = __builtin_bit_cast(int, x);
  yi = __builtin_amdgcn_update_dpp(0, xi, 0x140, 0xF, 0xF, true); // row mirror
  x  = fmaxf(x, __builtin_bit_cast(float, yi));
  return x;
}

// ---------------------------------------------------------------------------
// Kernel 0: convert w_attn [256,128] f32 -> f16
// ---------------------------------------------------------------------------
__global__ void k_cvt(const float* __restrict__ w, _Float16* __restrict__ o, int n) {
  int i = blockIdx.x * blockDim.x + threadIdx.x;
  if (i < n) o[i] = (_Float16)w[i];
}

// ---------------------------------------------------------------------------
// Kernel 1: fused projections theta/phi/g + 2x2 maxpool, f16 outputs.
// One wave per 2x2 quad of positions. Lane l owns out-channels {j*32+l}.
//   Q16 [B][N][32]   (position-major)     -> WMMA A fragments
//   K16 [B][N4][32]  (position-major)     -> WMMA B fragments (Q*K^T)
//   Vt16[B][128][N4] (channel-major)      -> WMMA B fragments (P*V)
// x addresses are wave-uniform -> readfirstlane so they become SMEM loads.
// ---------------------------------------------------------------------------
__global__ __launch_bounds__(256) void k_proj(
    const float* __restrict__ x,
    const float* __restrict__ w_theta, const float* __restrict__ b_theta,
    const float* __restrict__ w_phi,   const float* __restrict__ b_phi,
    const float* __restrict__ w_g,     const float* __restrict__ b_g,
    _Float16* __restrict__ Q16, _Float16* __restrict__ K16,
    _Float16* __restrict__ Vt16)
{
  const int lane = threadIdx.x & 31;
  const int wv   = threadIdx.x >> 5;
  const int gw   = blockIdx.x * 8 + wv;      // 0 .. 16383 (B * N4 quads)
  const int b    = __builtin_amdgcn_readfirstlane(gw >> 10);
  const int quad = __builtin_amdgcn_readfirstlane(gw & 1023);
  const int qr   = quad >> 5, qc = quad & 31;
  const int n00  = (2 * qr) * W_ + 2 * qc;   // wave-uniform

  float th[4] = {0.f, 0.f, 0.f, 0.f};
  float ph[4] = {0.f, 0.f, 0.f, 0.f};
  float gg[4][4] = {};

  const float* xb = x + (size_t)b * C_ * N_;
  for (int ch = 0; ch < C_; ++ch) {
    const float* xc = xb + (size_t)ch * N_;
    const float x0 = xc[n00],      x1 = xc[n00 + 1];
    const float x2 = xc[n00 + W_], x3 = xc[n00 + W_ + 1];
    const float wt = w_theta[lane * C_ + ch];
    const float wp = w_phi  [lane * C_ + ch];
    th[0] = fmaf(wt, x0, th[0]); th[1] = fmaf(wt, x1, th[1]);
    th[2] = fmaf(wt, x2, th[2]); th[3] = fmaf(wt, x3, th[3]);
    ph[0] = fmaf(wp, x0, ph[0]); ph[1] = fmaf(wp, x1, ph[1]);
    ph[2] = fmaf(wp, x2, ph[2]); ph[3] = fmaf(wp, x3, ph[3]);
#pragma unroll
    for (int j = 0; j < 4; ++j) {
      const float wg = w_g[(j * 32 + lane) * C_ + ch];
      gg[j][0] = fmaf(wg, x0, gg[j][0]); gg[j][1] = fmaf(wg, x1, gg[j][1]);
      gg[j][2] = fmaf(wg, x2, gg[j][2]); gg[j][3] = fmaf(wg, x3, gg[j][3]);
    }
  }

  const int nidx[4] = { n00, n00 + 1, n00 + W_, n00 + W_ + 1 };
  const float bt = b_theta[lane];
#pragma unroll
  for (int p = 0; p < 4; ++p)
    Q16[((size_t)b * N_ + nidx[p]) * CQ_ + lane] = (_Float16)(th[p] + bt);

  const int m = qr * 32 + qc;   // pooled position
  const float phv = fmaxf(fmaxf(ph[0], ph[1]), fmaxf(ph[2], ph[3])) + b_phi[lane];
  K16[((size_t)b * N4_ + m) * CQ_ + lane] = (_Float16)phv;

#pragma unroll
  for (int j = 0; j < 4; ++j) {
    const float gv = fmaxf(fmaxf(gg[j][0], gg[j][1]), fmaxf(gg[j][2], gg[j][3]))
                     + b_g[j * 32 + lane];
    Vt16[((size_t)b * CV_ + (j * 32 + lane)) * N4_ + m] = (_Float16)gv;
  }
}

// ---------------------------------------------------------------------------
// Kernel 2: fused flash attention. One wave = one 16-query tile.
// Per iteration processes 64 keys:
//   S = 4x WMMA(Q,K^T); row-max via DPP butterfly (no LDS traffic);
//   P -> LDS (C-layout) -> reload as two A-fragments;
//   row-sum of exp via WMMA(P, ones)  (no shuffles);
//   O += 16x WMMA(P, V).
// ---------------------------------------------------------------------------
__global__ __launch_bounds__(256) void k_attn(
    const _Float16* __restrict__ Q16, const _Float16* __restrict__ K16,
    const _Float16* __restrict__ Vt16, _Float16* __restrict__ AG16)
{
  __shared__ _Float16 ptile[8][16 * 64];   // 2KB per wave
  const int lane = threadIdx.x & 31;
  const int wv   = threadIdx.x >> 5;
  const int gt   = blockIdx.x * 8 + wv;    // 0 .. 4095 (B * N/16)
  const int b    = gt >> 8;
  const int qt   = gt & 255;
  const int hl   = lane >> 4;              // lane half (A/B K-split, C M-split)
  const int l16  = lane & 15;
  const int off  = hl ? 16 : 0;

  // Q A-fragment: lane row = l16, K chunks {off..off+7, off+8..off+15}
  AV qa;
  {
    const _Float16* qrow = Q16 + ((size_t)b * N_ + qt * 16 + l16) * CQ_;
    qa.h[0] = *(const v8h*)(qrow + off);
    qa.h[1] = *(const v8h*)(qrow + off + 8);
  }

  // all-ones B fragment for WMMA row sums
  AV ones;
#pragma unroll
  for (int i = 0; i < 16; ++i) ones.v[i] = (_Float16)1.0f;

  const v8f zero = {0.f, 0.f, 0.f, 0.f, 0.f, 0.f, 0.f, 0.f};
  v8f O[8];
#pragma unroll
  for (int i = 0; i < 8; ++i) O[i] = zero;
  float mrow[8], lrow[8];
#pragma unroll
  for (int r = 0; r < 8; ++r) { mrow[r] = -1e30f; lrow[r] = 0.f; }

  for (int kt4 = 0; kt4 < N4_ / 64; ++kt4) {
    const size_t kbase = (size_t)b * N4_ + kt4 * 64;

    // prefetch next iteration's K rows into cache (global_prefetch_b8)
    if (kt4 + 1 < N4_ / 64)
      __builtin_prefetch(K16 + (kbase + 64 + 2 * lane) * CQ_, 0, 0);

    // ---- S = Q * K^T for four adjacent 16-key tiles ----
    v8f s[4];
#pragma unroll
    for (int t = 0; t < 4; ++t) {
      AV kb;
      const _Float16* kr = K16 + (kbase + t * 16 + l16) * CQ_ + hl * 16;
      kb.v = *(const v16h*)kr;
      s[t] = wmma_f16(qa, kb, zero);
    }

    // ---- online max per row (DPP butterfly); exp; rescale O ----
#pragma unroll
    for (int r = 0; r < 8; ++r) {
      float t0 = fmaxf(fmaxf(s[0][r], s[1][r]), fmaxf(s[2][r], s[3][r]));
      t0 = wave16_max(t0);
      const float mn = fmaxf(mrow[r], t0);
      const float sc = __expf(mrow[r] - mn);
      mrow[r] = mn;
#pragma unroll
      for (int t = 0; t < 4; ++t) s[t][r] = __expf(s[t][r] - mn);
      lrow[r] *= sc;
#pragma unroll
      for (int cb = 0; cb < 8; ++cb) O[cb][r] *= sc;
    }

    // ---- P: C-layout regs -> LDS -> two A-fragments (keys 0..31, 32..63) ----
    _Float16* pt = &ptile[wv][0];
#pragma unroll
    for (int r = 0; r < 8; ++r) {
      const int row = r + hl * 8;
#pragma unroll
      for (int t = 0; t < 4; ++t)
        pt[row * 64 + t * 16 + l16] = (_Float16)s[t][r];
    }
    __syncthreads();
    AV pa0, pa1;
    {
      const _Float16* prow = pt + l16 * 64;
      pa0.h[0] = *(const v8h*)(prow + off);
      pa0.h[1] = *(const v8h*)(prow + off + 8);
      pa1.h[0] = *(const v8h*)(prow + 32 + off);
      pa1.h[1] = *(const v8h*)(prow + 32 + off + 8);
    }
    __syncthreads();

    // ---- row sums of exp via WMMA with all-ones B (replicated per row) ----
    v8f rs = wmma_f16(pa0, ones, zero);
    rs = wmma_f16(pa1, ones, rs);
#pragma unroll
    for (int r = 0; r < 8; ++r) lrow[r] += rs[r];

    // ---- O += P * V over 8 column tiles of V ----
#pragma unroll
    for (int cb = 0; cb < 8; ++cb) {
      AV vb0, vb1;
      const _Float16* vr =
          Vt16 + ((size_t)b * CV_ + cb * 16 + l16) * N4_ + kt4 * 64;
      vb0.v = *(const v16h*)(vr + hl * 16);
      vb1.v = *(const v16h*)(vr + 32 + hl * 16);
      O[cb] = wmma_f16(pa0, vb0, O[cb]);
      O[cb] = wmma_f16(pa1, vb1, O[cb]);
    }
  }

  // ---- normalize and store attn_g tile as f16 [B][N][128] ----
#pragma unroll
  for (int r = 0; r < 8; ++r) {
    const float inv = 1.0f / lrow[r];
    const int row = r + hl * 8;
    const int n = qt * 16 + row;
#pragma unroll
    for (int cb = 0; cb < 8; ++cb)
      AG16[((size_t)b * N_ + n) * CV_ + cb * 16 + l16] =
          (_Float16)(O[cb][r] * inv);
  }
}

// ---------------------------------------------------------------------------
// Kernel 3: final 1x1 conv (128->256) via WMMA + bias + residual.
// One wave = one [16 pos x 16 oc] tile, K = 128 (4 WMMAs).
// Epilogue: per lane the 8 accumulator rows are 8 consecutive n positions ->
// one aligned v8f residual load + one v8f store per lane.
// ---------------------------------------------------------------------------
__global__ __launch_bounds__(256) void k_out(
    const float* __restrict__ x, const _Float16* __restrict__ AG16,
    const _Float16* __restrict__ WA16, const float* __restrict__ b_attn,
    const float* __restrict__ sigma, float* __restrict__ out)
{
  const int lane = threadIdx.x & 31;
  const int wv   = threadIdx.x >> 5;
  const int gt   = blockIdx.x * 8 + wv;    // 0 .. 65535
  const int b    = gt >> 12;
  const int rem  = gt & 4095;
  const int qt   = rem >> 4;
  const int ot   = rem & 15;
  const int hl   = lane >> 4;
  const int l16  = lane & 15;
  const int off  = hl ? 16 : 0;

  const v8f zero = {0.f, 0.f, 0.f, 0.f, 0.f, 0.f, 0.f, 0.f};
  v8f acc = zero;
#pragma unroll
  for (int k = 0; k < 4; ++k) {
    AV a;
    const _Float16* ar =
        AG16 + ((size_t)b * N_ + qt * 16 + l16) * CV_ + k * 32;
    a.h[0] = *(const v8h*)(ar + off);
    a.h[1] = *(const v8h*)(ar + off + 8);
    AV w;
    const _Float16* wr =
        WA16 + (size_t)(ot * 16 + l16) * CV_ + k * 32 + hl * 16;
    w.v = *(const v16h*)wr;
    acc = wmma_f16(a, w, acc);
  }

  const float sg = sigma[0];
  const int oc = ot * 16 + l16;
  const float ba = b_attn[oc];

  // rows r..r+7 of this lane are consecutive n positions
  const int nbase = qt * 16 + hl * 8;
  const size_t base = ((size_t)(b * C_ + oc)) * N_ + nbase;
  const v8f xv = *(const v8f*)(x + base);
  v8f ov;
#pragma unroll
  for (int r = 0; r < 8; ++r) ov[r] = xv[r] + sg * (acc[r] + ba);
  *(v8f*)(out + base) = ov;
}

// ---------------------------------------------------------------------------
extern "C" void kernel_launch(void* const* d_in, const int* in_sizes, int n_in,
                              void* d_out, int out_size, void* d_ws, size_t ws_size,
                              hipStream_t stream) {
  const float* x       = (const float*)d_in[0];
  const float* w_theta = (const float*)d_in[1];
  const float* b_theta = (const float*)d_in[2];
  const float* w_phi   = (const float*)d_in[3];
  const float* b_phi   = (const float*)d_in[4];
  const float* w_g     = (const float*)d_in[5];
  const float* b_g     = (const float*)d_in[6];
  const float* w_attn  = (const float*)d_in[7];
  const float* b_attn  = (const float*)d_in[8];
  const float* sigma   = (const float*)d_in[9];
  float* out = (float*)d_out;

  char* ws = (char*)d_ws;
  _Float16* Q16  = (_Float16*)(ws);                       //  4 MB [B][N][32]
  _Float16* K16  = (_Float16*)(ws + 4194304);             //  1 MB [B][N4][32]
  _Float16* Vt16 = (_Float16*)(ws + 5242880);             //  4 MB [B][128][N4]
  _Float16* AG16 = (_Float16*)(ws + 9437184);             // 16 MB [B][N][128]
  _Float16* WA16 = (_Float16*)(ws + 26214400);            // 64 KB [256][128]

  k_cvt<<<(C_ * CV_ + 255) / 256, 256, 0, stream>>>(w_attn, WA16, C_ * CV_);
  k_proj<<<(B_ * N4_) / 8, 256, 0, stream>>>(x, w_theta, b_theta, w_phi, b_phi,
                                             w_g, b_g, Q16, K16, Vt16);
  k_attn<<<(B_ * (N_ / 16)) / 8, 256, 0, stream>>>(Q16, K16, Vt16, AG16);
  k_out<<<(B_ * (N_ / 16) * (C_ / 16)) / 8, 256, 0, stream>>>(
      x, AG16, WA16, b_attn, sigma, out);
}
